// MetricPatchEmbed_11321533792416
// MI455X (gfx1250) — matrix operand order repaired and checked by reading, not compiled, not run on hardware
//
#include <hip/hip_runtime.h>
#include <hip/hip_bf16.h>
#include <math.h>

// ---------------------------------------------------------------------------
// MetricPatchEmbed for MI455X (gfx1250, wave32, WMMA).
//   out[M=2048][C=768] = A[M][K=12544] * W[K][C] + proj_b,   K = kl*256 + v
// A: <=196 nonzeros/row (4 bilinear corners x 49 kernel positions),
//    scatter-built per 256-wide K group into LDS.
// W: proj_w repacked to bf16 in *WMMA-B-fragment order* Wb[q][n][kk]
//    (q=K/32 chunk, kk=K%32), so each lane's B fragment is 32 contiguous
//    bytes in global memory -> 2x global_load_b128, no LDS staging, no
//    inner-loop barriers. 19.3 MB, L2-resident.
// ---------------------------------------------------------------------------

#define IMGD 112
#define VV   256
#define CC   768
#define NPATCH 2048          // B(8) * 16 * 16
#define KTOT  (49 * 256)     // 12544

typedef __attribute__((ext_vector_type(16))) __bf16 v16bf;
typedef __attribute__((ext_vector_type(8)))  float  v8f;

static __device__ __forceinline__ unsigned short f32_bf16(float f) {
  unsigned int u = __float_as_uint(f);
  u += 0x7FFFu + ((u >> 16) & 1u);          // round-to-nearest-even
  return (unsigned short)(u >> 16);
}
static __device__ __forceinline__ float sigmoidf_(float x) {
  return 1.0f / (1.0f + __expf(-x));
}

// ---- Kernel A: per-patch conv params (one-hot conv == gather sum) ----------
__global__ void patch_params_kernel(const int* __restrict__ x,
                                    const float* __restrict__ mw,
                                    const float* __restrict__ mb,
                                    float* __restrict__ pp) {
  __shared__ float red[64 * 8];
  int p = blockIdx.x, t = threadIdx.x;
  int b = p >> 8, rem = p & 255, pi = rem >> 4, pj = rem & 15;
  float s[7] = {0.f,0.f,0.f,0.f,0.f,0.f,0.f};
  if (t < 49) {
    int pr = t / 7, pc = t % 7;
    int y = pi * 7 + pr, xx = pj * 7 + pc;
    int val = x[b * (IMGD * IMGD) + y * IMGD + xx];
    const float* wp = mw + (size_t)val * 49 + t;   // mw[o][val][t]
#pragma unroll
    for (int o = 0; o < 7; ++o) s[o] = wp[(size_t)o * (VV * 49)];
  }
#pragma unroll
  for (int o = 0; o < 7; ++o) red[t * 8 + o] = s[o];
  __syncthreads();
  for (int str = 32; str > 0; str >>= 1) {
    if (t < str) {
#pragma unroll
      for (int o = 0; o < 7; ++o) red[t * 8 + o] += red[(t + str) * 8 + o];
    }
    __syncthreads();
  }
  if (t == 0) {
    float par[7];
#pragma unroll
    for (int o = 0; o < 7; ++o) par[o] = red[o] + mb[o];
    float n = sqrtf(par[0] * par[0] + par[1] * par[1]);
    float inv = 1.0f / fmaxf(n, 1e-12f);
    float v0 = par[0] * inv, v1 = par[1] * inv;
    float sc = 0.5f + 1.5f * sigmoidf_(par[4]);
    float e0 = 2.0f * sigmoidf_(par[2]) * sc;
    float e1 = 2.0f * sigmoidf_(par[3]) * sc;
    // M = e0*v v^T + e1*u u^T, u = (-v1, v0)
    float m00 = e0 * v0 * v0 + e1 * v1 * v1;
    float m01 = (e0 - e1) * v0 * v1;
    float m11 = e0 * v1 * v1 + e1 * v0 * v0;
    float w0 = par[5], w1 = par[6];
    float wn = sqrtf(w0 * w0 + w1 * w1);
    float f = 0.5f * sigmoidf_(wn);            // (1-EPS_W)*sigmoid(|w|)
    float* o = pp + (size_t)p * 5;
    o[0] = m00; o[1] = m01; o[2] = m11; o[3] = w0 * f; o[4] = w1 * f;
  }
}

// ---- Kernel B: per (patch, kl) bilinear corners (value, weight) ------------
__global__ void corners_kernel(const int* __restrict__ x,
                               const float* __restrict__ pp,
                               int* __restrict__ cidx,
                               float* __restrict__ cwt) {
  int p = blockIdx.x, t = threadIdx.x;
  if (t >= 49) return;
  int b = p >> 8, rem = p & 255, pi = rem >> 4, pj = rem & 15;
  const float* q = pp + (size_t)p * 5;
  float m00 = q[0], m01 = q[1], m11 = q[2], w0 = q[3], w1 = q[4];
  int k = t / 7, l = t % 7;
  float dy = (float)(k - 3), dx = (float)(l - 3);
  float gn = sqrtf(dy * dy + dx * dx);
  float gi = 1.0f / fmaxf(gn, 1e-12f);
  float uy = dy * gi, ux = dx * gi;
  float quad = uy * uy * m00 + 2.0f * uy * ux * m01 + ux * ux * m11;
  float Fr = sqrtf(fmaxf(quad, 1e-12f)) + uy * w0 + ux * w1;
  Fr = fmaxf(Fr, 0.001f);
  float py = dy / Fr + (float)(pi * 7 + 3);
  float px = dx / Fr + (float)(pj * 7 + 3);
  float fy = floorf(py), fx = floorf(px);
  float wy = py - fy, wx = px - fx;
  int y0 = (int)fy, x0 = (int)fx;
  size_t base = ((size_t)p * 49 + t) * 4;
#pragma unroll
  for (int cy = 0; cy < 2; ++cy)
#pragma unroll
    for (int cx = 0; cx < 2; ++cx) {
      int yy = y0 + cy, xx = x0 + cx;
      bool valid = (yy >= 0) && (yy < IMGD) && (xx >= 0) && (xx < IMGD);
      float wq = (cy ? wy : 1.0f - wy) * (cx ? wx : 1.0f - wx);
      int vv = 0; float wo = 0.0f;
      if (valid) { vv = x[b * (IMGD * IMGD) + yy * IMGD + xx]; wo = wq; }
      cidx[base + cy * 2 + cx] = vv;
      cwt [base + cy * 2 + cx] = wo;
    }
}

// ---- Kernel C: repack proj_w (C,V,7,7) f32 -> Wb bf16, B-fragment order ----
// Wb[((q*768 + c)*32 + kk)], q = kl*8 + v/32, kk = v%32.
// Reads: each thread streams 49 contiguous floats (coalesced across threads).
// Writes: 32 consecutive threads (same v-group) write 64 contiguous bytes.
__global__ void pack_wb_kernel(const float* __restrict__ pw,
                               unsigned short* __restrict__ wb) {
  int c = blockIdx.x;                   // 0..767
  int v = threadIdx.x;                  // 0..255
  const float* src = pw + (size_t)c * KTOT + (size_t)v * 49;
  int qh = v >> 5, kk = v & 31;
  float vals[49];
#pragma unroll
  for (int kl = 0; kl < 49; ++kl) vals[kl] = src[kl];
#pragma unroll
  for (int kl = 0; kl < 49; ++kl) {
    size_t q = (size_t)(kl * 8 + qh);
    wb[(q * CC + c) * 32 + kk] = f32_bf16(vals[kl]);
  }
}

// ---- Kernel D: scatter-A bf16 WMMA GEMM ------------------------------------
// Tile: M=16 patches, N=256 cols; 256 threads = 8 waves; each wave owns two
// 16-wide N subtiles (2 WMMAs per A fragment). K: 49 groups of 256, inner
// chunks of 32 -> v_wmma_f32_16x16x32_bf16, f32 accumulation. B fragments
// come straight from global (pre-swizzled Wb), so the kc loop has no barriers.
#define AS 264   // A_s row stride in u16 (256 + pad)

__global__ void __launch_bounds__(256)
gemm_kernel(const int* __restrict__ cidx, const float* __restrict__ cwt,
            const unsigned short* __restrict__ wb,
            const float* __restrict__ pb, float* __restrict__ out) {
  __shared__ unsigned short A_s[16 * AS];
  int tid = threadIdx.x;
  int M0 = blockIdx.x * 16;
  int N0 = blockIdx.y * 256;
  int wv = tid >> 5, lane = tid & 31, half = lane >> 4, nl = lane & 15;

  // per-lane global byte offsets of the two B fragments within a q-chunk
  int n0 = N0 + wv * 32 + nl;                     // subtile 0 column
  const char* wbB = (const char*)wb;
  size_t off0 = ((size_t)n0 * 32 + (size_t)half * 16) * 2;
  size_t off1 = off0 + 16 * 32 * 2;               // subtile 1: n0+16

  v8f acc0 = {}, acc1 = {};
  for (int kl = 0; kl < 49; ++kl) {
    // zero A tile
    unsigned int* A32 = (unsigned int*)A_s;
    for (int i = tid; i < (16 * AS) / 2; i += 256) A32[i] = 0u;
    __syncthreads();
    // scatter: one thread per patch row, 4 corners, combine duplicate values
    if (tid < 16) {
      size_t base = ((size_t)(M0 + tid) * 49 + kl) * 4;
      int4   v4 = *(const int4*)(cidx + base);
      float4 w4 = *(const float4*)(cwt + base);
      int   vv[4] = {v4.x, v4.y, v4.z, v4.w};
      float ww[4] = {w4.x, w4.y, w4.z, w4.w};
#pragma unroll
      for (int c = 0; c < 4; ++c) {
        bool dup = false;
        for (int j = 0; j < c; ++j) dup |= (vv[j] == vv[c]);
        if (!dup) {
          float a = ww[c];
          for (int j = c + 1; j < 4; ++j) if (vv[j] == vv[c]) a += ww[j];
          A_s[tid * AS + vv[c]] = f32_bf16(a);
        }
      }
    }
    __syncthreads();

#pragma unroll
    for (int kc = 0; kc < 8; ++kc) {
      int q = kl * 8 + kc;
      const char* pB = wbB + (size_t)q * (CC * 64);   // 768 cols * 32 elem * 2B

      // B fragments: 32 contiguous bytes each, coalesced across the wave
      union { uint4 u4[2]; v16bf v; } b0, b1;
      b0.u4[0] = *(const uint4*)(pB + off0);
      b0.u4[1] = *(const uint4*)(pB + off0 + 16);
      b1.u4[0] = *(const uint4*)(pB + off1);
      b1.u4[1] = *(const uint4*)(pB + off1 + 16);

      // A fragment from LDS: lane M=nl; K=2j+16*half (contiguous, matches B)
      union { unsigned int u[8]; v16bf v; } af;
      {
        const unsigned short* ar = &A_s[nl * AS + kc * 32 + half * 16];
#pragma unroll
        for (int j = 0; j < 4; ++j) af.u[j] = *(const unsigned int*)(ar + 2 * j);
#pragma unroll
        for (int j = 0; j < 4; ++j) af.u[4 + j] = *(const unsigned int*)(ar + 8 + 2 * j);
      }

      acc0 = __builtin_amdgcn_wmma_f32_16x16x32_bf16(false, af.v, false, b0.v,
                                                     (short)0, acc0, false, false);
      acc1 = __builtin_amdgcn_wmma_f32_16x16x32_bf16(false, af.v, false, b1.v,
                                                     (short)0, acc1, false, false);
    }
    __syncthreads();   // protect A_s before next group's zero+scatter
  }

  // epilogue: D layout -> M = r + 8*half, N = nl within each subtile
  int n = N0 + wv * 32 + nl;
  float pb0 = pb[n], pb1 = pb[n + 16];
#pragma unroll
  for (int r = 0; r < 8; ++r) {
    int m = half * 8 + r;
    out[(size_t)(M0 + m) * CC + n]      = acc0[r] + pb0;
    out[(size_t)(M0 + m) * CC + n + 16] = acc1[r] + pb1;
  }
}

// ---------------------------------------------------------------------------
extern "C" void kernel_launch(void* const* d_in, const int* in_sizes, int n_in,
                              void* d_out, int out_size, void* d_ws, size_t ws_size,
                              hipStream_t stream) {
  const int*   x  = (const int*)d_in[0];
  const float* mw = (const float*)d_in[1];
  const float* mb = (const float*)d_in[2];
  const float* pw = (const float*)d_in[3];
  const float* pb = (const float*)d_in[4];
  float* out = (float*)d_out;

  char* ws = (char*)d_ws;
  // workspace layout (256B aligned):
  //   pp   : 2048*5 f32               =    40,960 B @ 0
  //   cidx : 2048*49*4 i32            = 1,605,632 B @ 40,960
  //   cwt  : 2048*49*4 f32            = 1,605,632 B @ 1,646,592
  //   Wb   : 12544*768 bf16 (swizzled)=19,267,584 B @ 3,252,224
  float*          pp   = (float*)(ws + 0);
  int*            cidx = (int*)  (ws + 40960);
  float*          cwt  = (float*)(ws + 1646592);
  unsigned short* wb   = (unsigned short*)(ws + 3252224);

  patch_params_kernel<<<NPATCH, 64, 0, stream>>>(x, mw, mb, pp);
  corners_kernel<<<NPATCH, 64, 0, stream>>>(x, pp, cidx, cwt);
  pack_wb_kernel<<<CC, 256, 0, stream>>>(pw, wb);
  gemm_kernel<<<dim3(NPATCH / 16, CC / 256), 256, 0, stream>>>(cidx, cwt, wb, pb, out);
}
